// EncoderRoPEBlock_18433999635094
// MI455X (gfx1250) — matrix operand order repaired and checked by reading, not compiled, not run on hardware
//
#include <hip/hip_runtime.h>
#include <hip/hip_bf16.h>

// ---------------------------------------------------------------------------
// Problem constants (reference: B=2, L=2048, D=1024, H=16, hd=64, Dff=4096)
// ---------------------------------------------------------------------------
#define BB   2
#define LSEQ 2048
#define DMOD 1024
#define NH   16
#define HD   64
#define DFF  4096
#define MROWS (BB * LSEQ)          // 4096 token rows

typedef __bf16 bf16t;
typedef bf16t v16bf __attribute__((ext_vector_type(16)));
typedef float v8f   __attribute__((ext_vector_type(8)));

// bf16 stored as raw u16 everywhere; convert explicitly (round-to-nearest-even)
__device__ __forceinline__ unsigned short f2bf(float f) {
    unsigned int u = __float_as_uint(f);
    unsigned int r = u + 0x7FFFu + ((u >> 16) & 1u);
    return (unsigned short)(r >> 16);
}

struct Frag { union { v16bf v; uint4 q[2]; }; };

__device__ __forceinline__ v8f wmma_bf16(const Frag& a, const Frag& b, v8f c) {
    return __builtin_amdgcn_wmma_f32_16x16x32_bf16(
        false, a.v, false, b.v, (short)0, c, false, false);
}

// CDNA5 async global->LDS DMA (ASYNCcnt-tracked, bypasses VGPRs)
__device__ __forceinline__ void async_ld16(unsigned lds_off, const void* gptr) {
    asm volatile("global_load_async_to_lds_b128 %0, %1, off"
                 :: "v"(lds_off), "v"((unsigned long long)gptr) : "memory");
}
__device__ __forceinline__ void wait_async0() {
    asm volatile("s_wait_asynccnt 0x0" ::: "memory");
}

// ---------------------------------------------------------------------------
// Weight prep: W[K][N] f32  ->  WT[N][K] bf16 (tiled transpose through LDS)
// ---------------------------------------------------------------------------
__global__ __launch_bounds__(256) void transpose_cvt(
    const float* __restrict__ W, unsigned short* __restrict__ WT, int Kd, int Nd)
{
    __shared__ float tile[32][33];
    const int kb = blockIdx.x * 32, nb = blockIdx.y * 32;
    const int tx = threadIdx.x & 31, ty = threadIdx.x >> 5;   // ty in 0..7
    #pragma unroll
    for (int i = 0; i < 32; i += 8)
        tile[ty + i][tx] = W[(size_t)(kb + ty + i) * Nd + nb + tx];
    __syncthreads();
    #pragma unroll
    for (int i = 0; i < 32; i += 8)
        WT[(size_t)(nb + ty + i) * Kd + kb + tx] = f2bf(tile[tx][ty + i]);
}

// ---------------------------------------------------------------------------
// LayerNorm over D=1024, one row per 256-thread block, bf16 output
// ---------------------------------------------------------------------------
__global__ __launch_bounds__(256) void ln_kernel(
    const float* __restrict__ x, const float* __restrict__ g,
    const float* __restrict__ b, unsigned short* __restrict__ out)
{
    __shared__ float red[256];
    const int row = blockIdx.x;
    const float* xr = x + (size_t)row * DMOD;
    float v[4]; float s = 0.f;
    #pragma unroll
    for (int i = 0; i < 4; ++i) { v[i] = xr[threadIdx.x + i * 256]; s += v[i]; }
    red[threadIdx.x] = s; __syncthreads();
    for (int off = 128; off > 0; off >>= 1) {
        if (threadIdx.x < off) red[threadIdx.x] += red[threadIdx.x + off];
        __syncthreads();
    }
    const float mean = red[0] * (1.f / DMOD);
    __syncthreads();
    float vs = 0.f;
    #pragma unroll
    for (int i = 0; i < 4; ++i) { float d = v[i] - mean; vs += d * d; }
    red[threadIdx.x] = vs; __syncthreads();
    for (int off = 128; off > 0; off >>= 1) {
        if (threadIdx.x < off) red[threadIdx.x] += red[threadIdx.x + off];
        __syncthreads();
    }
    const float rstd = rsqrtf(red[0] * (1.f / DMOD) + 1e-6f);
    #pragma unroll
    for (int i = 0; i < 4; ++i) {
        const int c = threadIdx.x + i * 256;
        out[(size_t)row * DMOD + c] = f2bf((v[i] - mean) * rstd * g[c] + b[c]);
    }
}

// ---------------------------------------------------------------------------
// WMMA GEMM: C = A[M][K](bf16) * BT[N][K](bf16) + bias, fused epilogues.
// Block tile 128x128, 8 waves, each wave = 4x2 grid of 16x16 WMMA tiles.
// Staging: double-buffered LDS filled by async global->LDS DMA.
// ---------------------------------------------------------------------------
enum { EPI_QROPE = 0, EPI_KROPE = 1, EPI_VT = 2, EPI_OPROJ = 3,
       EPI_GELU = 4, EPI_MLP2 = 5 };

template <int EPI>
__global__ __launch_bounds__(256) void gemm_wmma(
    const unsigned short* __restrict__ A,   // [M][K]
    const unsigned short* __restrict__ BT,  // [N][K]
    const float* __restrict__ bias,         // [N]
    int M, int N, int K,
    unsigned short* __restrict__ outB,
    float* __restrict__ outF,
    const float* __restrict__ res,
    const float* __restrict__ gate)
{
    constexpr int TM = 128, TN = 128, TK = 32;
    constexpr int LDR = 80;                      // padded LDS row pitch (bytes)
    constexpr int BUFSZ = (TM + TN) * LDR;       // one double-buffer stage
    __shared__ __align__(16) unsigned char smem[2 * BUFSZ];

    const int tid  = threadIdx.x;
    const int lane = tid & 31;
    const int wave = tid >> 5;
    const int half = lane >> 4;
    const int l16  = lane & 15;
    const int m0 = blockIdx.x * TM;
    const int n0 = blockIdx.y * TN;
    const int waveM = (wave >> 2) * 64;          // 2 wave-rows
    const int waveN = (wave & 3) * 32;           // 4 wave-cols

    const unsigned smem_off = (unsigned)(unsigned long long)(void*)smem; // LDS byte offset
    const int srow = tid >> 1, sq2 = (tid & 1) * 2;  // 2 chunks/row/thread

    // issue one K-step worth of async DMA into buffer `boff`
    auto stage = [&](int k0, unsigned boff) {
        #pragma unroll
        for (int q = 0; q < 2; ++q) {            // 2 x b128 per matrix per thread
            async_ld16(smem_off + boff + srow * LDR + (sq2 + q) * 16,
                       A + (size_t)(m0 + srow) * K + k0 + (sq2 + q) * 8);
            async_ld16(smem_off + boff + TM * LDR + srow * LDR + (sq2 + q) * 16,
                       BT + (size_t)(n0 + srow) * K + k0 + (sq2 + q) * 8);
        }
    };

    v8f acc[4][2];
    #pragma unroll
    for (int i = 0; i < 4; ++i)
        #pragma unroll
        for (int j = 0; j < 2; ++j)
            #pragma unroll
            for (int r = 0; r < 8; ++r) acc[i][j][r] = 0.f;

    // prologue: fill buffer 0
    stage(0, 0);
    wait_async0();
    __syncthreads();

    int buf = 0;
    for (int k0 = 0; k0 < K; k0 += TK) {
        const int nbuf = buf ^ 1;
        if (k0 + TK < K) stage(k0 + TK, (unsigned)(nbuf * BUFSZ)); // overlap DMA

        const unsigned char* sA = smem + buf * BUFSZ;
        const unsigned char* sB = sA + TM * LDR;

        Frag af[4], bf[2];
        #pragma unroll
        for (int mi = 0; mi < 4; ++mi) {
            const unsigned char* rp = sA + (waveM + mi * 16 + l16) * LDR;
            af[mi].q[0] = *reinterpret_cast<const uint4*>(rp + 32 * half);      // K = 16h..16h+7
            af[mi].q[1] = *reinterpret_cast<const uint4*>(rp + 32 * half + 16); // K = 16h+8..
        }
        #pragma unroll
        for (int ni = 0; ni < 2; ++ni) {
            const unsigned char* rp = sB + (waveN + ni * 16 + l16) * LDR;
            bf[ni].q[0] = *reinterpret_cast<const uint4*>(rp + 32 * half);      // K = 16h..16h+7
            bf[ni].q[1] = *reinterpret_cast<const uint4*>(rp + 32 * half + 16); // K = 16h+8..
        }
        #pragma unroll
        for (int mi = 0; mi < 4; ++mi)
            #pragma unroll
            for (int ni = 0; ni < 2; ++ni)
                acc[mi][ni] = wmma_bf16(af[mi], bf[ni], acc[mi][ni]);

        wait_async0();          // our DMA into nbuf is done
        __syncthreads();        // everyone's DMA visible; reads of buf finished
        buf = nbuf;
    }

    // ---- epilogues (C layout: row = r + 8*half, col = lane&15) ----
    #pragma unroll
    for (int mi = 0; mi < 4; ++mi)
        #pragma unroll
        for (int ni = 0; ni < 2; ++ni) {
            const int gn = n0 + waveN + ni * 16 + l16;
            const float bv = bias ? bias[gn] : 0.f;
            #pragma unroll
            for (int r = 0; r < 8; ++r) {
                const int gm = m0 + waveM + mi * 16 + r + 8 * half;
                float y = acc[mi][ni][r] + bv;
                if (EPI == EPI_QROPE || EPI == EPI_KROPE) {
                    // RoPE: adjacent column lives in the adjacent lane
                    const float partner = __shfl_xor(y, 1, 32);
                    const int bidx = gm >> 11, l = gm & (LSEQ - 1);
                    const int h = gn >> 6, c = gn & (HD - 1);
                    const int i = c >> 1;
                    const float ang = (float)l * __expf(-0.28782314f * (float)i);
                    float sn, cs; __sincosf(ang, &sn, &cs);
                    const float o = (c & 1) ? (y * cs + partner * sn)
                                            : (y * cs - partner * sn);
                    outB[((size_t)((bidx * NH + h) * LSEQ + l)) * HD + c] = f2bf(o);
                } else if (EPI == EPI_VT) {
                    const int bidx = gm >> 11, l = gm & (LSEQ - 1);
                    const int h = gn >> 6, c = gn & (HD - 1);
                    outB[((size_t)((bidx * NH + h) * HD + c)) * LSEQ + l] = f2bf(y);
                } else if (EPI == EPI_OPROJ) {
                    const size_t idx = (size_t)gm * DMOD + gn;
                    outF[idx] = res[idx] + gate[gn] * y;
                } else if (EPI == EPI_GELU) {
                    const float g = 0.5f * y * (1.f + erff(y * 0.70710678f));
                    outB[(size_t)gm * N + gn] = f2bf(g);
                } else { // EPI_MLP2
                    const size_t idx = (size_t)gm * DMOD + gn;
                    outF[idx] = outF[idx] + gate[gn] * y;
                }
            }
        }
}

// ---------------------------------------------------------------------------
// Flash attention: one wave per 16 query rows; online softmax; WMMA for
// both Q*K^T and P*V. q/k: [B*H][L][64] bf16 (RoPE'd), vt: [B*H][64][L] bf16.
// ---------------------------------------------------------------------------
__global__ __launch_bounds__(128) void attn_kernel(
    const unsigned short* __restrict__ q,
    const unsigned short* __restrict__ k,
    const unsigned short* __restrict__ vt,
    unsigned short* __restrict__ out)       // [B*L][D] bf16
{
    __shared__ __align__(16) unsigned char psmem[4 * 16 * 80]; // per-wave P tile
    const int lane = threadIdx.x & 31;
    const int wave = threadIdx.x >> 5;
    const int half = lane >> 4;
    const int l16  = lane & 15;
    const int bh = blockIdx.x >> 5;         // 32 q-tiles of 64 rows per head
    const int qt = blockIdx.x & 31;
    const int qbase = qt * 64 + wave * 16;
    const unsigned short* qp = q  + (size_t)bh * LSEQ * HD;
    const unsigned short* kp = k  + (size_t)bh * LSEQ * HD;
    const unsigned short* vp = vt + (size_t)bh * HD * LSEQ;
    unsigned char* ps = psmem + wave * 16 * 80;

    // Q A-fragments (rows qbase..qbase+15, K-steps of 32 over hd=64)
    Frag qa[2];
    #pragma unroll
    for (int ks = 0; ks < 2; ++ks) {
        const unsigned short* rp = qp + (size_t)(qbase + l16) * HD + ks * 32;
        qa[ks].q[0] = *reinterpret_cast<const uint4*>(rp + 16 * half);
        qa[ks].q[1] = *reinterpret_cast<const uint4*>(rp + 16 * half + 8);
    }

    float mrow[8], lrow[8];
    v8f o[4];
    #pragma unroll
    for (int r = 0; r < 8; ++r) { mrow[r] = -1e30f; lrow[r] = 0.f; }
    #pragma unroll
    for (int nt = 0; nt < 4; ++nt)
        #pragma unroll
        for (int r = 0; r < 8; ++r) o[nt][r] = 0.f;

    for (int j0 = 0; j0 < LSEQ; j0 += 32) {
        v8f s0, s1;
        #pragma unroll
        for (int r = 0; r < 8; ++r) { s0[r] = 0.f; s1[r] = 0.f; }
        #pragma unroll
        for (int ks = 0; ks < 2; ++ks) {   // K rows are already B^T layout
            Frag kb0, kb1;
            const unsigned short* r0 = kp + (size_t)(j0 + l16) * HD + ks * 32 + 16 * half;
            kb0.q[0] = *reinterpret_cast<const uint4*>(r0);
            kb0.q[1] = *reinterpret_cast<const uint4*>(r0 + 8);
            const unsigned short* r1 = kp + (size_t)(j0 + 16 + l16) * HD + ks * 32 + 16 * half;
            kb1.q[0] = *reinterpret_cast<const uint4*>(r1);
            kb1.q[1] = *reinterpret_cast<const uint4*>(r1 + 8);
            s0 = wmma_bf16(qa[ks], kb0, s0);
            s1 = wmma_bf16(qa[ks], kb1, s1);
        }
        // online softmax (per-row stats; rows replicate across a 16-lane half)
        float nm[8], alpha[8];
        #pragma unroll
        for (int r = 0; r < 8; ++r) {
            s0[r] *= 0.125f; s1[r] *= 0.125f;            // 1/sqrt(64)
            float t = fmaxf(s0[r], s1[r]);
            #pragma unroll
            for (int off = 1; off < 16; off <<= 1) t = fmaxf(t, __shfl_xor(t, off, 32));
            nm[r] = fmaxf(mrow[r], t);
        }
        #pragma unroll
        for (int r = 0; r < 8; ++r) {
            s0[r] = __expf(s0[r] - nm[r]);
            s1[r] = __expf(s1[r] - nm[r]);
            alpha[r] = __expf(mrow[r] - nm[r]);
            mrow[r] = nm[r];
            float t = s0[r] + s1[r];
            #pragma unroll
            for (int off = 1; off < 16; off <<= 1) t += __shfl_xor(t, off, 32);
            lrow[r] = lrow[r] * alpha[r] + t;
        }
        #pragma unroll
        for (int nt = 0; nt < 4; ++nt)
            #pragma unroll
            for (int r = 0; r < 8; ++r) o[nt][r] *= alpha[r];

        // C-layout P -> LDS (16x32 bf16 tile) -> A-layout fragment
        #pragma unroll
        for (int r = 0; r < 8; ++r) {
            *reinterpret_cast<unsigned short*>(ps + (r + 8 * half) * 80 + l16 * 2)      = f2bf(s0[r]);
            *reinterpret_cast<unsigned short*>(ps + (r + 8 * half) * 80 + 32 + l16 * 2) = f2bf(s1[r]);
        }
        __builtin_amdgcn_wave_barrier();
        asm volatile("s_wait_dscnt 0" ::: "memory");
        Frag pa;
        {
            const unsigned char* rp = ps + l16 * 80;
            pa.q[0] = *reinterpret_cast<const uint4*>(rp + 32 * half);
            pa.q[1] = *reinterpret_cast<const uint4*>(rp + 32 * half + 16);
        }
        __builtin_amdgcn_wave_barrier();
        // P * V  (V pre-transposed -> contiguous B-fragment loads)
        #pragma unroll
        for (int nt = 0; nt < 4; ++nt) {
            Frag vb;
            const unsigned short* vr = vp + (size_t)(nt * 16 + l16) * LSEQ + j0 + 16 * half;
            vb.q[0] = *reinterpret_cast<const uint4*>(vr);
            vb.q[1] = *reinterpret_cast<const uint4*>(vr + 8);
            o[nt] = wmma_bf16(pa, vb, o[nt]);
        }
    }

    const int bidx = bh >> 4, h = bh & 15;
    #pragma unroll
    for (int r = 0; r < 8; ++r) {
        const float inv = 1.f / lrow[r];
        const int gl = qbase + r + 8 * half;
        const size_t rowbase = ((size_t)(bidx * LSEQ + gl)) * DMOD + h * HD;
        #pragma unroll
        for (int nt = 0; nt < 4; ++nt)
            out[rowbase + nt * 16 + l16] = f2bf(o[nt][r] * inv);
    }
}

// ---------------------------------------------------------------------------
// Host launcher
// ---------------------------------------------------------------------------
extern "C" void kernel_launch(void* const* d_in, const int* in_sizes, int n_in,
                              void* d_out, int out_size, void* d_ws, size_t ws_size,
                              hipStream_t stream)
{
    const float* x     = (const float*)d_in[0];
    const float* ln1_g = (const float*)d_in[1];
    const float* ln1_b = (const float*)d_in[2];
    const float* wq    = (const float*)d_in[3];
    const float* bq    = (const float*)d_in[4];
    const float* wk    = (const float*)d_in[5];
    const float* bk    = (const float*)d_in[6];
    const float* wv    = (const float*)d_in[7];
    const float* bv    = (const float*)d_in[8];
    const float* wo    = (const float*)d_in[9];
    const float* bo    = (const float*)d_in[10];
    const float* ln2_g = (const float*)d_in[11];
    const float* ln2_b = (const float*)d_in[12];
    const float* w1    = (const float*)d_in[13];
    const float* b1    = (const float*)d_in[14];
    const float* w2    = (const float*)d_in[15];
    const float* b2    = (const float*)d_in[16];
    const float* gmsa  = (const float*)d_in[17];
    const float* gmlp  = (const float*)d_in[18];
    float* out = (float*)d_out;

    unsigned char* ws = (unsigned char*)d_ws;
    size_t off = 0;
    auto alloc = [&](size_t bytes) -> void* {
        void* p = ws + off; off += (bytes + 255) & ~(size_t)255; return p;
    };
    unsigned short* hb   = (unsigned short*)alloc((size_t)MROWS * DMOD * 2);
    unsigned short* wqT  = (unsigned short*)alloc((size_t)DMOD * DMOD * 2);
    unsigned short* wkT  = (unsigned short*)alloc((size_t)DMOD * DMOD * 2);
    unsigned short* wvT  = (unsigned short*)alloc((size_t)DMOD * DMOD * 2);
    unsigned short* woT  = (unsigned short*)alloc((size_t)DMOD * DMOD * 2);
    unsigned short* w1T  = (unsigned short*)alloc((size_t)DFF * DMOD * 2);
    unsigned short* w2T  = (unsigned short*)alloc((size_t)DMOD * DFF * 2);
    unsigned short* qr   = (unsigned short*)alloc((size_t)BB * NH * LSEQ * HD * 2);
    unsigned short* kr   = (unsigned short*)alloc((size_t)BB * NH * LSEQ * HD * 2);
    unsigned short* vtb  = (unsigned short*)alloc((size_t)BB * NH * HD * LSEQ * 2);
    unsigned short* aout = (unsigned short*)alloc((size_t)MROWS * DMOD * 2);
    unsigned short* h2b  = (unsigned short*)alloc((size_t)MROWS * DMOD * 2);
    unsigned short* mlp1 = (unsigned short*)alloc((size_t)MROWS * DFF * 2);
    (void)ws_size; (void)in_sizes; (void)n_in; (void)out_size;

    // 1) weights -> bf16 [N][K]
    transpose_cvt<<<dim3(DMOD / 32, DMOD / 32), 256, 0, stream>>>(wq, wqT, DMOD, DMOD);
    transpose_cvt<<<dim3(DMOD / 32, DMOD / 32), 256, 0, stream>>>(wk, wkT, DMOD, DMOD);
    transpose_cvt<<<dim3(DMOD / 32, DMOD / 32), 256, 0, stream>>>(wv, wvT, DMOD, DMOD);
    transpose_cvt<<<dim3(DMOD / 32, DMOD / 32), 256, 0, stream>>>(wo, woT, DMOD, DMOD);
    transpose_cvt<<<dim3(DMOD / 32, DFF / 32),  256, 0, stream>>>(w1, w1T, DMOD, DFF);
    transpose_cvt<<<dim3(DFF / 32,  DMOD / 32), 256, 0, stream>>>(w2, w2T, DFF, DMOD);

    // 2) LN1 -> bf16
    ln_kernel<<<MROWS, 256, 0, stream>>>(x, ln1_g, ln1_b, hb);

    // 3) QKV projections (fused RoPE / V-transpose epilogues)
    dim3 gQKV(MROWS / 128, DMOD / 128);
    gemm_wmma<EPI_QROPE><<<gQKV, 256, 0, stream>>>(hb, wqT, bq, MROWS, DMOD, DMOD,
                                                   qr, nullptr, nullptr, nullptr);
    gemm_wmma<EPI_KROPE><<<gQKV, 256, 0, stream>>>(hb, wkT, bk, MROWS, DMOD, DMOD,
                                                   kr, nullptr, nullptr, nullptr);
    gemm_wmma<EPI_VT><<<gQKV, 256, 0, stream>>>(hb, wvT, bv, MROWS, DMOD, DMOD,
                                                vtb, nullptr, nullptr, nullptr);

    // 4) flash attention
    attn_kernel<<<BB * NH * (LSEQ / 64), 128, 0, stream>>>(qr, kr, vtb, aout);

    // 5) O-proj + gated residual -> x1 (f32, stored in d_out)
    gemm_wmma<EPI_OPROJ><<<gQKV, 256, 0, stream>>>(aout, woT, bo, MROWS, DMOD, DMOD,
                                                   nullptr, out, x, gmsa);

    // 6) LN2 -> bf16
    ln_kernel<<<MROWS, 256, 0, stream>>>(out, ln2_g, ln2_b, h2b);

    // 7) MLP up + exact GELU
    gemm_wmma<EPI_GELU><<<dim3(MROWS / 128, DFF / 128), 256, 0, stream>>>(
        h2b, w1T, b1, MROWS, DFF, DMOD, mlp1, nullptr, nullptr, nullptr);

    // 8) MLP down + gated residual (in-place on d_out)
    gemm_wmma<EPI_MLP2><<<gQKV, 256, 0, stream>>>(mlp1, w2T, b2, MROWS, DMOD, DFF,
                                                  nullptr, out, nullptr, gmlp);
}
